// RevealModel_43482248905418
// MI455X (gfx1250) — matrix operand order repaired and checked by reading, not compiled, not run on hardware
//
#include <hip/hip_runtime.h>

// ---------------- problem constants ----------------
constexpr int NN = 50000;   // nodes (3125 * 16)
constexpr int NE = 800000;  // edges
constexpr int IN = 128;     // input feature dim
constexpr int H  = 256;     // hidden
constexpr int H3 = 768;     // 3*H
constexpr int NL = 6;       // GGNN layers
constexpr int NG = 512;     // graphs
constexpr int NC = 2;       // classes

// ---------------- types ----------------
typedef __attribute__((ext_vector_type(16))) __bf16        v16bf;
typedef __attribute__((ext_vector_type(8)))  float         v8f;
typedef __attribute__((ext_vector_type(4)))  unsigned int  v4u;

union ABfrag { v4u q[2]; v16bf v; };   // 32B: one WMMA 16x32 bf16 operand per lane

__device__ __forceinline__ v8f v8f_zero() {
  v8f z = {0.f,0.f,0.f,0.f,0.f,0.f,0.f,0.f};
  return z;
}

__device__ __forceinline__ unsigned short f2bf(float f) {
  unsigned u = __float_as_uint(f);
  u += 0x7FFFu + ((u >> 16) & 1u);          // round-to-nearest-even
  return (unsigned short)(u >> 16);
}

__device__ __forceinline__ v8f wmma_bf16(v16bf a, v16bf b, v8f c) {
  // D = A(16x32 bf16) x B(32x16 bf16) + C(16x16 f32)
  return __builtin_amdgcn_wmma_f32_16x16x32_bf16(false, a, false, b, (short)0, c,
                                                 false, false);
}

// ---------------- weight packing: fp32 -> bf16 WMMA B-fragments ----------------
// Fragment (kt, nt) of B[K,Nc]: 32 lanes x 8 dwords, stored at
//   dword offset ((nt*KT + kt)*32 + lane)*8 + v
// Lane semantics (ISA 7.12.2, 16-bit B 32x16): col = nt*16 + (lane&15),
//   lane half h holds K = kt*32 + 16*(v>>2) + 8h + 2*(v&3), +1.
__global__ void pack_b_kernel(const float* __restrict__ src, unsigned int* __restrict__ dst,
                              int K, int Nc, int trans, int ld) {
  const int t = blockIdx.x * blockDim.x + threadIdx.x;
  const int total = (K * Nc) >> 1;
  if (t >= total) return;
  const int KT   = K >> 5;
  const int f    = t >> 8;
  const int lane = (t >> 3) & 31;
  const int v    = t & 7;
  const int nt   = f / KT;
  const int kt   = f - nt * KT;
  const int col  = nt * 16 + (lane & 15);
  const int kb   = kt * 32 + ((v >> 2) << 4) + ((lane >> 4) << 3) + 2 * (v & 3);
  float e0, e1;
  if (trans) { // B[k,j] = src[j*ld + k]   (for X @ W^T)
    e0 = src[(size_t)col * ld + kb];
    e1 = src[(size_t)col * ld + kb + 1];
  } else {     // B[k,j] = src[k*ld + j]   (for X @ W)
    e0 = src[(size_t)kb * ld + col];
    e1 = src[(size_t)(kb + 1) * ld + col];
  }
  dst[t] = (unsigned)f2bf(e0) | ((unsigned)f2bf(e1) << 16);
}

// ---------------- generic WMMA GEMM:  C[M,Nc] = A_bf16[M,K] @ Bpack (+bias, relu) ----
// KT is compile-time (K = 32*KT) so the fully-unrolled K-loop addresses every A/B
// fragment with immediate offsets from ONE base pointer each (no per-iter VALU math).
// block = 256 threads = 8 waves; wave (rh, cg): rows 16*rh, cols 64*cg -> 32x256/block
template <int KT>
__global__ void gemm_bf16_wmma(const unsigned short* __restrict__ Abf,
                               const unsigned int*  __restrict__ Bpack,
                               const float* __restrict__ bias,
                               float* __restrict__ Cf,
                               unsigned short* __restrict__ Cbf,
                               int M, int Nc, int relu) {
  constexpr int K = KT * 32;
  const int lane = threadIdx.x & 31;
  const int wave = threadIdx.x >> 5;
  const int rh = wave >> 2, cg = wave & 3;
  const int row0 = blockIdx.x * 32 + rh * 16;
  if (row0 >= M) return;                       // wave-uniform guard (EXEC stays all-1)
  const int colBase = blockIdx.y * 256 + cg * 64;
  const int g = lane >> 4, lm = lane & 15;

  // single base pointers; all loop offsets become instruction immediates
  const unsigned short* aBase = Abf + (size_t)(row0 + lm) * K + 8 * g;
  const unsigned int*   bBase = Bpack + (size_t)(colBase >> 4) * KT * 256 + lane * 8;

  v8f acc[4];
#pragma unroll
  for (int t = 0; t < 4; ++t) acc[t] = v8f_zero();

#pragma unroll
  for (int kt = 0; kt < KT; ++kt) {
    ABfrag a;
    a.q[0] = *(const v4u*)(aBase + kt * 32);        // K pairs 0..7 / 8..15 (lane group)
    a.q[1] = *(const v4u*)(aBase + kt * 32 + 16);   //        16..23 / 24..31
#pragma unroll
    for (int t = 0; t < 4; ++t) {
      const v4u* bp = (const v4u*)(bBase + (t * KT + kt) * 256);
      ABfrag b;
      b.q[0] = bp[0];
      b.q[1] = bp[1];
      acc[t] = wmma_bf16(a.v, b.v, acc[t]);
    }
  }

#pragma unroll
  for (int t = 0; t < 4; ++t) {
    const int col = colBase + t * 16 + lm;
    const float bv = bias ? bias[col] : 0.0f;
    float* crow = Cf + (size_t)(row0 + g * 8) * Nc + col;
#pragma unroll
    for (int i = 0; i < 8; ++i) {                    // C/D: VGPR i -> row i (+8 hi lanes)
      float val = acc[t][i] + bv;
      if (relu) val = fmaxf(val, 0.0f);
      crow[(size_t)i * Nc] = val;
    }
    if (Cbf) {
      unsigned short* cbrow = Cbf + (size_t)(row0 + g * 8) * Nc + col;
#pragma unroll
      for (int i = 0; i < 8; ++i) cbrow[(size_t)i * Nc] = f2bf(acc[t][i] + bv > 0.f || !relu
                                                              ? (relu ? fmaxf(acc[t][i] + bv, 0.f)
                                                                      : acc[t][i] + bv)
                                                              : 0.f);
    }
  }
}

// ---------------- fused GRU: gi = aggr@Wih^T+bih ; gh = h@Whh^T+bhh ; gates ---------
// block = 16 rows, 8 waves; wave w owns H-columns [w*32, w*32+32) of all three gate
// segments for BOTH gi and gh  -> 12 accumulator tiles, gates computed in registers.
// All B offsets are immediates off one wave-dependent base (s*128 + ct*8 + kt tiles).
__global__ void gru_fused_wmma(const unsigned short* __restrict__ aggrBf,
                               unsigned short* __restrict__ hBf,   // in/out (row-private)
                               float* __restrict__ hF,             // in/out
                               const unsigned int* __restrict__ WihP,
                               const unsigned int* __restrict__ WhhP,
                               const float* __restrict__ bih,
                               const float* __restrict__ bhh) {
  const int lane = threadIdx.x & 31;
  const int wave = threadIdx.x >> 5;
  const int g = lane >> 4, lm = lane & 15;
  const int row0 = blockIdx.x * 16;
  const unsigned short* aBase = aggrBf + (size_t)(row0 + lm) * H + 8 * g;
  const unsigned short* hBase = hBf    + (size_t)(row0 + lm) * H + 8 * g;
  // frag (s,ct,kt): dword offset = ((s*16 + wave*2 + ct)*8 + kt)*256 + lane*8
  const unsigned int* biBase = WihP + (size_t)wave * 16 * 256 + lane * 8;
  const unsigned int* bhBase = WhhP + (size_t)wave * 16 * 256 + lane * 8;

  v8f accI[3][2], accH[3][2];
#pragma unroll
  for (int s = 0; s < 3; ++s)
#pragma unroll
    for (int ct = 0; ct < 2; ++ct) { accI[s][ct] = v8f_zero(); accH[s][ct] = v8f_zero(); }

#pragma unroll
  for (int kt = 0; kt < 8; ++kt) {               // K = 256 = 8 * 32
    ABfrag aA, aH;
    aA.q[0] = *(const v4u*)(aBase + kt * 32);
    aA.q[1] = *(const v4u*)(aBase + kt * 32 + 16);
    aH.q[0] = *(const v4u*)(hBase + kt * 32);
    aH.q[1] = *(const v4u*)(hBase + kt * 32 + 16);
#pragma unroll
    for (int s = 0; s < 3; ++s) {
#pragma unroll
      for (int ct = 0; ct < 2; ++ct) {
        const int fo = (s * 128 + ct * 8 + kt) * 256;   // immediate-expressible
        ABfrag b;
        const v4u* bi = (const v4u*)(biBase + fo);
        b.q[0] = bi[0]; b.q[1] = bi[1];
        accI[s][ct] = wmma_bf16(aA.v, b.v, accI[s][ct]);
        const v4u* bh = (const v4u*)(bhBase + fo);
        b.q[0] = bh[0]; b.q[1] = bh[1];
        accH[s][ct] = wmma_bf16(aH.v, b.v, accH[s][ct]);
      }
    }
  }

#pragma unroll
  for (int ct = 0; ct < 2; ++ct) {
    const int c = wave * 32 + ct * 16 + lm;      // column within H
    const float bi0 = bih[c], bi1 = bih[H + c], bi2 = bih[2 * H + c];
    const float bh0 = bhh[c], bh1 = bhh[H + c], bh2 = bhh[2 * H + c];
    float*          hrow  = hF  + (size_t)(row0 + g * 8) * H + c;
    unsigned short* hbrow = hBf + (size_t)(row0 + g * 8) * H + c;
#pragma unroll
    for (int i = 0; i < 8; ++i) {
      const float gr = (accI[0][ct][i] + bi0) + (accH[0][ct][i] + bh0);
      const float gz = (accI[1][ct][i] + bi1) + (accH[1][ct][i] + bh1);
      const float r  = 1.0f / (1.0f + __expf(-gr));
      const float z  = 1.0f / (1.0f + __expf(-gz));
      const float nn = tanhf((accI[2][ct][i] + bi2) + r * (accH[2][ct][i] + bh2));
      const float hold = hrow[(size_t)i * H];
      const float hn = (1.0f - z) * nn + z * hold;
      hrow [(size_t)i * H] = hn;
      hbrow[(size_t)i * H] = f2bf(hn);           // feeds next layer's WMMA A operand
    }
  }
}

// ---------------- small elementwise / scatter kernels ----------------
__global__ void init_h_kernel(const float* __restrict__ x, float* __restrict__ h,
                              unsigned short* __restrict__ hbf) {
  const int t = blockIdx.x * blockDim.x + threadIdx.x;    // exactly NN*H threads
  const int n = t >> 8, c = t & 255;
  const float v = (c < IN) ? x[(size_t)n * IN + c] : 0.0f;
  h[t] = v;
  hbf[t] = f2bf(v);
}

__global__ void zero_f32_kernel(float* __restrict__ p, int n) {
  const int t = blockIdx.x * blockDim.x + threadIdx.x;
  if (t < n) p[t] = 0.0f;
}

__global__ void f32_to_bf16_kernel(const float* __restrict__ s,
                                   unsigned short* __restrict__ d, int n) {
  const int t = blockIdx.x * blockDim.x + threadIdx.x;
  if (t < n) d[t] = f2bf(s[t]);
}

__global__ void scatter_add_edges(const int* __restrict__ ei,
                                  const float* __restrict__ m,
                                  float* __restrict__ aggr) {
  const int e = blockIdx.x;            // one block per edge, 256 ch coalesced
  const int s = ei[e];
  const int d = ei[NE + e];
  const int c = threadIdx.x;
  unsafeAtomicAdd(&aggr[(size_t)d * H + c], m[(size_t)s * H + c]);
}

__global__ void pool_add_kernel(const float* __restrict__ h,
                                const int* __restrict__ batch,
                                float* __restrict__ gsum) {
  const int t = blockIdx.x * blockDim.x + threadIdx.x;    // NN*H threads
  const int n = t >> 8, c = t & 255;
  unsafeAtomicAdd(&gsum[(size_t)batch[n] * H + c], h[t]);
}

__global__ void classifier_kernel(const float* __restrict__ g3,
                                  const float* __restrict__ w,
                                  const float* __restrict__ b,
                                  float* __restrict__ out) {
  const int t = blockIdx.x * blockDim.x + threadIdx.x;
  if (t >= NG * NC) return;
  const int gi = t >> 1, c = t & 1;
  float s = b[c];
  const float* gr = g3 + (size_t)gi * 512;
  const float* wr = w + (size_t)c * 512;
  for (int k = 0; k < 512; ++k) s += gr[k] * wr[k];
  out[t] = s;
}

// ---------------- host driver ----------------
extern "C" void kernel_launch(void* const* d_in, const int* in_sizes, int n_in,
                              void* d_out, int out_size, void* d_ws, size_t ws_size,
                              hipStream_t stream) {
  (void)in_sizes; (void)n_in; (void)out_size; (void)ws_size;
  const float* x     = (const float*)d_in[0];
  const int*   ei    = (const int*)  d_in[1];
  const int*   batch = (const int*)  d_in[2];
  const float* ggnnW = (const float*)d_in[3];
  const float* w_ih  = (const float*)d_in[4];
  const float* w_hh  = (const float*)d_in[5];
  const float* b_ih  = (const float*)d_in[6];
  const float* b_hh  = (const float*)d_in[7];
  const float* w1    = (const float*)d_in[8];
  const float* b1    = (const float*)d_in[9];
  const float* w2    = (const float*)d_in[10];
  const float* b2    = (const float*)d_in[11];
  const float* w3    = (const float*)d_in[12];
  const float* b3    = (const float*)d_in[13];
  const float* cw    = (const float*)d_in[14];
  const float* cb    = (const float*)d_in[15];
  float* out = (float*)d_out;

  char* ws = (char*)d_ws;
  size_t off = 0;
  auto take = [&](size_t bytes) -> char* {
    char* p = ws + off;
    off += (bytes + 255) & ~(size_t)255;
    return p;
  };
  float*          h      = (float*)         take((size_t)NN * H * 4);
  unsigned short* hbf    = (unsigned short*)take((size_t)NN * H * 2);
  float*          m      = (float*)         take((size_t)NN * H * 4);
  float*          aggr   = (float*)         take((size_t)NN * H * 4);
  unsigned short* aggrbf = (unsigned short*)take((size_t)NN * H * 2);
  unsigned int*   packW  = (unsigned int*)  take((size_t)NL * H * H * 2);
  unsigned int*   packIh = (unsigned int*)  take((size_t)H * H3 * 2);
  unsigned int*   packHh = (unsigned int*)  take((size_t)H * H3 * 2);
  unsigned int*   packM1 = (unsigned int*)  take((size_t)H * 512 * 2);
  unsigned int*   packM2 = (unsigned int*)  take((size_t)512 * H * 2);
  unsigned int*   packM3 = (unsigned int*)  take((size_t)H * 512 * 2);
  float*          gpool  = (float*)         take((size_t)NG * H * 4);
  unsigned short* gpoolbf= (unsigned short*)take((size_t)NG * H * 2);
  float*          g1     = (float*)         take((size_t)NG * 512 * 4);
  unsigned short* g1bf   = (unsigned short*)take((size_t)NG * 512 * 2);
  float*          g2     = (float*)         take((size_t)NG * H * 4);
  unsigned short* g2bf   = (unsigned short*)take((size_t)NG * H * 2);
  float*          g3     = (float*)         take((size_t)NG * 512 * 4);

  // ---- pack all weights into WMMA bf16 B-fragment layout (deterministic, each call)
  for (int l = 0; l < NL; ++l)
    pack_b_kernel<<<(H * H / 2 + 255) / 256, 256, 0, stream>>>(
        ggnnW + (size_t)l * H * H, packW + (size_t)l * H * H / 2, H, H, 0, H);
  pack_b_kernel<<<(H * H3 / 2 + 255) / 256, 256, 0, stream>>>(w_ih, packIh, H, H3, 1, H);
  pack_b_kernel<<<(H * H3 / 2 + 255) / 256, 256, 0, stream>>>(w_hh, packHh, H, H3, 1, H);
  pack_b_kernel<<<(H * 512 / 2 + 255) / 256, 256, 0, stream>>>(w1, packM1, H, 512, 1, H);
  pack_b_kernel<<<(512 * H / 2 + 255) / 256, 256, 0, stream>>>(w2, packM2, 512, H, 1, 512);
  pack_b_kernel<<<(H * 512 / 2 + 255) / 256, 256, 0, stream>>>(w3, packM3, H, 512, 1, H);

  // ---- h = pad(x) ----
  init_h_kernel<<<NN, 256, 0, stream>>>(x, h, hbf);

  // ---- 6 GGNN layers ----
  for (int l = 0; l < NL; ++l) {
    gemm_bf16_wmma<8><<<dim3((NN + 31) / 32, 1), 256, 0, stream>>>(
        hbf, packW + (size_t)l * H * H / 2, nullptr, m, nullptr, NN, H, 0);
    zero_f32_kernel<<<NN, 256, 0, stream>>>(aggr, NN * H);
    scatter_add_edges<<<NE, 256, 0, stream>>>(ei, m, aggr);
    f32_to_bf16_kernel<<<NN, 256, 0, stream>>>(aggr, aggrbf, NN * H);
    gru_fused_wmma<<<NN / 16, 256, 0, stream>>>(aggrbf, hbf, h, packIh, packHh, b_ih, b_hh);
  }

  // ---- global_add_pool + MLP + classifier ----
  zero_f32_kernel<<<(NG * H + 255) / 256, 256, 0, stream>>>(gpool, NG * H);
  pool_add_kernel<<<NN, 256, 0, stream>>>(h, batch, gpool);
  f32_to_bf16_kernel<<<(NG * H + 255) / 256, 256, 0, stream>>>(gpool, gpoolbf, NG * H);
  gemm_bf16_wmma<8> <<<dim3(NG / 32, 2), 256, 0, stream>>>(gpoolbf, packM1, b1, g1, g1bf, NG, 512, 1);
  gemm_bf16_wmma<16><<<dim3(NG / 32, 1), 256, 0, stream>>>(g1bf,    packM2, b2, g2, g2bf, NG, H,   1);
  gemm_bf16_wmma<8> <<<dim3(NG / 32, 2), 256, 0, stream>>>(g2bf,    packM3, b3, g3, nullptr, NG, 512, 1);
  classifier_kernel<<<(NG * NC + 255) / 256, 256, 0, stream>>>(g3, cw, cb, out);
}
